// MultiHeadAttention_91104846283325
// MI455X (gfx1250) — matrix operand order repaired and checked by reading, not compiled, not run on hardware
//
#include <hip/hip_runtime.h>
#include <hip/hip_bf16.h>

typedef __bf16 bf16;
typedef __attribute__((ext_vector_type(16))) __bf16 v16bf;
typedef __attribute__((ext_vector_type(8)))  __bf16 v8bf;
typedef __attribute__((ext_vector_type(8)))  float   v8f;

#define E_   1024
#define T_   2048
#define B_   2
#define NH_  16
#define HS_  64
#define M_   (B_ * T_)   // 4096 token rows

// ---------------- WMMA fragment loaders (bf16, 16x16x32) ----------------

// A fragment: 16x32 (MxK), row-major memory. `row` = full M row index for this lane.
// ISA layout: lanes 0-15 hold K = +0..7 & +16..23; lanes 16-31 hold K = +8..15 & +24..31.
static __device__ __forceinline__ v16bf ld_afrag(const bf16* __restrict__ base,
                                                 int ld, int row, int k0, int lane) {
  const bf16* p = base + (size_t)row * ld + k0 + ((lane & 16) ? 8 : 0);
  union { v16bf v; v8bf h[2]; } u;
  u.h[0] = *(const v8bf*)(p);
  u.h[1] = *(const v8bf*)(p + 16);
  return u.v;
}

// B fragment: 32x16 (KxN). Memory holds B^T: row `col` contiguous over K.
// ISA layout: lanes 0-15 hold K = +0..7 & +16..23; lanes 16-31 hold K = +8..15 & +24..31.
static __device__ __forceinline__ v16bf ld_bfrag(const bf16* __restrict__ base,
                                                 int ld, int col, int k0, int lane) {
  const bf16* p = base + (size_t)col * ld + k0 + ((lane & 16) ? 8 : 0);
  union { v16bf v; v8bf h[2]; } u;
  u.h[0] = *(const v8bf*)(p);
  u.h[1] = *(const v8bf*)(p + 16);
  return u.v;
}

#define WMMA_BF16(a, b, c) \
  __builtin_amdgcn_wmma_f32_16x16x32_bf16(false, (a), false, (b), (short)0, (c), false, false)

// ---------------- fp32 -> bf16 conversion ----------------

__global__ void __launch_bounds__(256) cvt_bf16_kernel(const float* __restrict__ in,
                                                       bf16* __restrict__ out, int n) {
  int i = blockIdx.x * blockDim.x + threadIdx.x;
  int stride = gridDim.x * blockDim.x;
  for (; i < n; i += stride) out[i] = (bf16)in[i];
}

// ---------------- QKV projection: y = x @ W^T + bias (scaled) ----------------
// Per-wave tile: 32M x 64N (2 A-frags x 4 B-frags -> 8 WMMAs per k-chunk).
// mode 0: out[b][h][t][d]  (Q, K)    mode 1: out[b][h][d][t]  (V transposed)

__global__ void __launch_bounds__(256) proj_kernel(const bf16* __restrict__ X,
                                                   const bf16* __restrict__ W,
                                                   const float* __restrict__ bias,
                                                   bf16* __restrict__ out,
                                                   float scale, int mode) {
  const int lane = threadIdx.x & 31;
  const int w    = blockIdx.x * 8 + (threadIdx.x >> 5);
  const int nt   = w & 15;          // E_/64 = 16 column supertiles
  const int mt   = w >> 4;          // M_/32 = 128 row supertiles

  const int arow0 = mt * 32 + (lane & 15);
  const int ncol0 = nt * 64 + (lane & 15);

  v8f acc[2][4] = {{{}, {}, {}, {}}, {{}, {}, {}, {}}};

#pragma unroll 2
  for (int k = 0; k < E_; k += 32) {
    v16bf a0 = ld_afrag(X, E_, arow0,      k, lane);
    v16bf a1 = ld_afrag(X, E_, arow0 + 16, k, lane);
#pragma unroll
    for (int j = 0; j < 4; ++j) {
      v16bf b = ld_bfrag(W, E_, ncol0 + j * 16, k, lane);
      acc[0][j] = WMMA_BF16(a0, b, acc[0][j]);
      acc[1][j] = WMMA_BF16(a1, b, acc[1][j]);
    }
  }

#pragma unroll
  for (int j = 0; j < 4; ++j) {
    const int   n  = ncol0 + j * 16;   // C/D layout: lane owns column N
    const float bb = bias[n];
    const int   hh = n >> 6, dd = n & 63;
#pragma unroll
    for (int i = 0; i < 2; ++i) {
#pragma unroll
      for (int r = 0; r < 8; ++r) {
        int   m  = mt * 32 + i * 16 + r + ((lane & 16) ? 8 : 0);
        int   bi = m >> 11;            // / T_
        int   t  = m & (T_ - 1);
        float v  = (acc[i][j][r] + bb) * scale;
        size_t off = (mode == 0)
            ? ((size_t)(bi * NH_ + hh) * T_ + t) * HS_ + dd
            : ((size_t)(bi * NH_ + hh) * HS_ + dd) * T_ + t;
        out[off] = (bf16)v;
      }
    }
  }
}

// ---------------- Flash attention: one wave per (b, h, 32-row q-tile) ----------
// Computes S^T = K·Q^T so S^T C/D registers alias directly onto the P
// A-fragment layout for the P·V WMMA. Two 16-row q-groups share the K-frag
// and V-frag loads (2x arithmetic intensity on K/V traffic).

__global__ void __launch_bounds__(256) attn_kernel(const bf16* __restrict__ Q,
                                                   const bf16* __restrict__ K,
                                                   const bf16* __restrict__ Vt,
                                                   bf16* __restrict__ A) {
  const int lane = threadIdx.x & 31;
  const int w    = blockIdx.x * 8 + (threadIdx.x >> 5);
  const int qt   = w & 63;                  // T_/32 = 64 q supertiles
  const int h    = (w >> 6) & (NH_ - 1);
  const int b    = w >> 10;
  const int qb   = qt * 32;

  const bf16* Qb = Q  + (size_t)(b * NH_ + h) * T_ * HS_;
  const bf16* Kb = K  + (size_t)(b * NH_ + h) * T_ * HS_;
  const bf16* Vb = Vt + (size_t)(b * NH_ + h) * HS_ * T_;

  // Q^T as B operand (column q = contiguous Q row): 2 q-groups x 2 d-chunks.
  v16bf bq[2][2];
#pragma unroll
  for (int g = 0; g < 2; ++g) {
    bq[g][0] = ld_bfrag(Qb, HS_, qb + g * 16 + (lane & 15), 0,  lane);
    bq[g][1] = ld_bfrag(Qb, HS_, qb + g * 16 + (lane & 15), 32, lane);
  }

  float m_run[2] = {-1e30f, -1e30f};
  float l_run[2] = {0.0f, 0.0f};
  v8f acc[2][4] = {{{}, {}, {}, {}}, {{}, {}, {}, {}}};

  const int koff = (lane & 16) ? 8 : 0;
  const int qmax = qb + 31;

  for (int kt = 0; kt <= qmax; kt += 32) {
    // K A-fragments, shared by both q-groups: 2 k-subtiles x 2 d-chunks.
    v16bf ak[2][2];
#pragma unroll
    for (int i = 0; i < 2; ++i) {
      ak[i][0] = ld_afrag(Kb, HS_, kt + i * 16 + (lane & 15), 0,  lane);
      ak[i][1] = ld_afrag(Kb, HS_, kt + i * 16 + (lane & 15), 32, lane);
    }
    // V B-fragments, shared by both q-groups: 4 d-tiles.
    v16bf bv[4];
#pragma unroll
    for (int n = 0; n < 4; ++n)
      bv[n] = ld_bfrag(Vb, T_, n * 16 + (lane & 15), kt, lane);

#pragma unroll
    for (int g = 0; g < 2; ++g) {
      const int qcol = qb + g * 16 + (lane & 15);   // this lane's q row
      // S^T tiles for this q-group
      v8f st0 = {}, st1 = {};
      st0 = WMMA_BF16(ak[0][0], bq[g][0], st0);
      st0 = WMMA_BF16(ak[0][1], bq[g][1], st0);
      st1 = WMMA_BF16(ak[1][0], bq[g][0], st1);
      st1 = WMMA_BF16(ak[1][1], bq[g][1], st1);

      // Causal mask + per-lane partial row max (lane pair L, L^16 shares a q row)
      float mloc = -1e30f;
#pragma unroll
      for (int r = 0; r < 8; ++r) {
        float s0 = (kt + koff + r      <= qcol) ? st0[r] : -1e30f;
        float s1 = (kt + 16 + koff + r <= qcol) ? st1[r] : -1e30f;
        st0[r] = s0; st1[r] = s1;
        mloc = fmaxf(mloc, fmaxf(s0, s1));
      }
      mloc = fmaxf(mloc, __shfl_xor(mloc, 16));     // full row max
      float m_new = fmaxf(m_run[g], mloc);
      float alpha = __expf(m_run[g] - m_new);
      m_run[g] = m_new;
      l_run[g] *= alpha;

      // Rescale O accumulator: alpha for row rr lives in lane rr.
#pragma unroll
      for (int r = 0; r < 8; ++r) {
        float ar = __shfl(alpha, r + ((lane & 16) >> 1));
        acc[g][0][r] *= ar; acc[g][1][r] *= ar;
        acc[g][2][r] *= ar; acc[g][3][r] *= ar;
      }

      // P = exp(S^T - m): S^T C/D regs map 1:1 onto the P A-fragment layout.
      union { v16bf v; __bf16 e[16]; } ap;
#pragma unroll
      for (int r = 0; r < 8; ++r) {
        float p0 = __expf(st0[r] - m_new);
        float p1 = __expf(st1[r] - m_new);
        l_run[g] += p0 + p1;
        ap.e[r]     = (__bf16)p0;
        ap.e[r + 8] = (__bf16)p1;
      }

      // O += P @ V (V frags shared between q-groups)
#pragma unroll
      for (int n = 0; n < 4; ++n)
        acc[g][n] = WMMA_BF16(ap.v, bv[n], acc[g][n]);
    }
  }

  bf16* Ab = A + (size_t)b * T_ * E_ + (size_t)h * HS_;
#pragma unroll
  for (int g = 0; g < 2; ++g) {
    float l = l_run[g] + __shfl_xor(l_run[g], 16);  // full row sum
    float inv = 1.0f / l;
#pragma unroll
    for (int r = 0; r < 8; ++r) {
      float ir = __shfl(inv, r + ((lane & 16) >> 1));
      int   t  = qb + g * 16 + r + ((lane & 16) ? 8 : 0);
#pragma unroll
      for (int n = 0; n < 4; ++n) {
        int d = n * 16 + (lane & 15);
        Ab[(size_t)t * E_ + d] = (bf16)(acc[g][n][r] * ir);
      }
    }
  }
}

// ---------------- Output projection: out = a @ Wo^T + bo (f32 out) ----------
// Same 32M x 64N per-wave tiling as proj_kernel.

__global__ void __launch_bounds__(256) outproj_kernel(const bf16* __restrict__ X,
                                                      const bf16* __restrict__ W,
                                                      const float* __restrict__ bias,
                                                      float* __restrict__ out) {
  const int lane = threadIdx.x & 31;
  const int w    = blockIdx.x * 8 + (threadIdx.x >> 5);
  const int nt   = w & 15;
  const int mt   = w >> 4;

  const int arow0 = mt * 32 + (lane & 15);
  const int ncol0 = nt * 64 + (lane & 15);

  v8f acc[2][4] = {{{}, {}, {}, {}}, {{}, {}, {}, {}}};

#pragma unroll 2
  for (int k = 0; k < E_; k += 32) {
    v16bf a0 = ld_afrag(X, E_, arow0,      k, lane);
    v16bf a1 = ld_afrag(X, E_, arow0 + 16, k, lane);
#pragma unroll
    for (int j = 0; j < 4; ++j) {
      v16bf b = ld_bfrag(W, E_, ncol0 + j * 16, k, lane);
      acc[0][j] = WMMA_BF16(a0, b, acc[0][j]);
      acc[1][j] = WMMA_BF16(a1, b, acc[1][j]);
    }
  }

#pragma unroll
  for (int j = 0; j < 4; ++j) {
    const int   n  = ncol0 + j * 16;
    const float bb = bias[n];
#pragma unroll
    for (int i = 0; i < 2; ++i) {
#pragma unroll
      for (int r = 0; r < 8; ++r) {
        int m = mt * 32 + i * 16 + r + ((lane & 16) ? 8 : 0);
        out[(size_t)m * E_ + n] = acc[i][j][r] + bb;
      }
    }
  }
}

// ---------------- host launcher ----------------

extern "C" void kernel_launch(void* const* d_in, const int* in_sizes, int n_in,
                              void* d_out, int out_size, void* d_ws, size_t ws_size,
                              hipStream_t stream) {
  (void)in_sizes; (void)n_in; (void)out_size; (void)ws_size;
  const float* x  = (const float*)d_in[0];
  const float* Wq = (const float*)d_in[1];
  const float* bq = (const float*)d_in[2];
  const float* Wk = (const float*)d_in[3];
  const float* bk = (const float*)d_in[4];
  const float* Wv = (const float*)d_in[5];
  const float* bv = (const float*)d_in[6];
  const float* Wo = (const float*)d_in[7];
  const float* bo = (const float*)d_in[8];
  float* out = (float*)d_out;

  char* ws = (char*)d_ws;
  const size_t tokBytes = (size_t)M_ * E_ * sizeof(bf16);   // 8 MB
  const size_t wBytes   = (size_t)E_ * E_ * sizeof(bf16);   // 2 MB
  bf16* xb  = (bf16*)ws;               ws += tokBytes;
  bf16* wqb = (bf16*)ws;               ws += wBytes;
  bf16* wkb = (bf16*)ws;               ws += wBytes;
  bf16* wvb = (bf16*)ws;               ws += wBytes;
  bf16* wob = (bf16*)ws;               ws += wBytes;
  bf16* Qh  = (bf16*)ws;               ws += tokBytes;      // [B,NH,T,HS]
  bf16* Kh  = (bf16*)ws;               ws += tokBytes;      // [B,NH,T,HS]
  bf16* Vt  = (bf16*)ws;               ws += tokBytes;      // [B,NH,HS,T]
  bf16* Ab  = (bf16*)ws;               ws += tokBytes;      // [B,T,E]

  // fp32 -> bf16
  cvt_bf16_kernel<<<dim3(1024), dim3(256), 0, stream>>>(x,  xb,  M_ * E_);
  cvt_bf16_kernel<<<dim3(256),  dim3(256), 0, stream>>>(Wq, wqb, E_ * E_);
  cvt_bf16_kernel<<<dim3(256),  dim3(256), 0, stream>>>(Wk, wkb, E_ * E_);
  cvt_bf16_kernel<<<dim3(256),  dim3(256), 0, stream>>>(Wv, wvb, E_ * E_);
  cvt_bf16_kernel<<<dim3(256),  dim3(256), 0, stream>>>(Wo, wob, E_ * E_);

  // QKV projections: 2048 wave-tiles each = 256 blocks of 8 waves
  const dim3 gProj(256), blk(256);
  proj_kernel<<<gProj, blk, 0, stream>>>(xb, wqb, bq, Qh, 0.125f, 0); // Q, pre-scaled 1/sqrt(HS)
  proj_kernel<<<gProj, blk, 0, stream>>>(xb, wkb, bk, Kh, 1.0f,   0); // K
  proj_kernel<<<gProj, blk, 0, stream>>>(xb, wvb, bv, Vt, 1.0f,   1); // V (transposed)

  // Attention: 2048 q-supertile waves = 256 blocks
  attn_kernel<<<dim3(256), blk, 0, stream>>>(Qh, Kh, Vt, Ab);

  // Output projection -> f32 d_out
  outproj_kernel<<<gProj, blk, 0, stream>>>(Ab, wob, bo, out);
}
